// JointEBM_27676769255477
// MI455X (gfx1250) — compile-verified, matrix operand order checked by reading, not run on hardware
//
#include <hip/hip_runtime.h>

// ---------------------------------------------------------------------------
// Joint-EBM inference-by-gradient-descent, fused single kernel for gfx1250.
// Each workgroup (256 thr = 8 waves) owns 128 batch rows; each wave owns a
// 16-row strip and runs all `steps` iterations locally using f16 WMMA with
// f32 accumulation. x-projection c = x@W1x + b1 is computed once and kept in
// registers as the WMMA C operand of the per-step z1 GEMM.
// ---------------------------------------------------------------------------

typedef __attribute__((ext_vector_type(16))) _Float16 v16h;
typedef __attribute__((ext_vector_type(8)))  _Float16 v8h;
typedef __attribute__((ext_vector_type(8)))  float    v8f;

#define LRATE 0.1f

// ----- dynamic LDS partition (bytes) -----
static constexpr int OFF_W1X  = 0;                  // 128x128 f16, [n][k]  (W1 rows 0..127 transposed)
static constexpr int OFF_W2B  = OFF_W1X + 32768;    // 128x128 f16, [n][k]  (W2 transposed)
static constexpr int OFF_W2T  = OFF_W2B + 32768;    // 128x128 f16, row-major W2 == [n][k] of W2^T
static constexpr int OFF_W1Y  = OFF_W2T + 32768;    // 128x32  f16, [n][kpad] (W1 rows 128..143, K padded 16->32)
static constexpr int OFF_W1YT = OFF_W1Y + 8192;     // 16x128  f16, [n][k]  (W1 rows 128..143 row-major)
static constexpr int OFF_STG  = OFF_W1YT + 4096;    // 8 waves x 16x128 f16 staging (h1 / gz2 / gz1)
static constexpr int OFF_YPAD = OFF_STG + 32768;    // 8 waves x 16x32  f16 y staging (K padded, cols 16..31 = 0)
static constexpr int SMEM_BYTES = OFF_YPAD + 8192;  // 151552 B = 148 KB

// Load a 16x32 f16 fragment (A layout; B uses identical addressing when the
// matrix is stored K-contiguous per column). base points at the 16-row/col
// strip; element (i, K) lives at base[i*strideH + K].
__device__ __forceinline__ v16h lds_frag(const _Float16* base, int strideH,
                                         int kbase, int l15, int lhi) {
  const _Float16* p = base + l15 * strideH + kbase + (lhi << 3);
  v8h lo = *(const v8h*)(p);        // K = kbase+koff .. +7
  v8h hi = *(const v8h*)(p + 16);   // K = kbase+16+koff .. +7
  return __builtin_shufflevector(lo, hi, 0, 1, 2, 3, 4, 5, 6, 7,
                                 8, 9, 10, 11, 12, 13, 14, 15);
}

// A fragment of x (f32 in global memory, row-major, 128 cols) -> f16 frag.
__device__ __forceinline__ v16h gmem_frag_x(const float* xrow, int kbase, int lhi) {
  const float* p = xrow + kbase + (lhi << 3);
  float4 a0 = *(const float4*)(p + 0);
  float4 a1 = *(const float4*)(p + 4);
  float4 c0 = *(const float4*)(p + 16);
  float4 c1 = *(const float4*)(p + 20);
  v16h r;
  r[0]  = (_Float16)a0.x; r[1]  = (_Float16)a0.y; r[2]  = (_Float16)a0.z; r[3]  = (_Float16)a0.w;
  r[4]  = (_Float16)a1.x; r[5]  = (_Float16)a1.y; r[6]  = (_Float16)a1.z; r[7]  = (_Float16)a1.w;
  r[8]  = (_Float16)c0.x; r[9]  = (_Float16)c0.y; r[10] = (_Float16)c0.z; r[11] = (_Float16)c0.w;
  r[12] = (_Float16)c1.x; r[13] = (_Float16)c1.y; r[14] = (_Float16)c1.z; r[15] = (_Float16)c1.w;
  return r;
}

#define WMMA_F16(A, B, C) \
  __builtin_amdgcn_wmma_f32_16x16x32_f16(false, (A), false, (B), (short)0, (C), false, false)

__global__ void __launch_bounds__(256, 1)
JointEBM_kernel(const float* __restrict__ x, const int* __restrict__ t,
                const float* __restrict__ W1, const float* __restrict__ b1,
                const float* __restrict__ W2, const float* __restrict__ b2,
                const float* __restrict__ W3, const int* __restrict__ stepsp,
                float* __restrict__ out) {
  extern __shared__ __align__(16) char smem[];
  _Float16* sW1x  = (_Float16*)(smem + OFF_W1X);
  _Float16* sW2b  = (_Float16*)(smem + OFF_W2B);
  _Float16* sW2t  = (_Float16*)(smem + OFF_W2T);
  _Float16* sW1y  = (_Float16*)(smem + OFF_W1Y);
  _Float16* sW1yT = (_Float16*)(smem + OFF_W1YT);
  _Float16* sStg  = (_Float16*)(smem + OFF_STG);
  _Float16* sYpad = (_Float16*)(smem + OFF_YPAD);

  const int tid  = threadIdx.x;
  const int wave = tid >> 5;
  const int lane = tid & 31;
  const int l15  = lane & 15;
  const int lhi  = lane >> 4;

  // ---- cooperative weight load / transpose into LDS (f32 -> f16) ----
  for (int i = tid; i < 128 * 128; i += 256) {
    int k = i >> 7, n = i & 127;
    sW1x[n * 128 + k] = (_Float16)W1[i];   // [n][k] of W1_x
    sW2b[n * 128 + k] = (_Float16)W2[i];   // [n][k] of W2
    sW2t[i]           = (_Float16)W2[i];   // row-major W2 == [n][k] of W2^T
  }
  for (int i = tid; i < 128 * 32; i += 256) {
    int n = i >> 5, k = i & 31;
    sW1y[i] = (k < 16) ? (_Float16)W1[(128 + k) * 128 + n] : (_Float16)0.f;
  }
  for (int i = tid; i < 16 * 128; i += 256)
    sW1yT[i] = (_Float16)W1[128 * 128 + i];          // [n][k] of W1_y^T
  for (int i = tid; i < 8 * 16 * 32; i += 256)
    sYpad[i] = (_Float16)0.f;                        // zero K-pad (cols 16..31 stay 0)
  __syncthreads();

  _Float16* stg  = sStg  + wave * (16 * 128);
  _Float16* ypad = sYpad + wave * (16 * 32);
  const long long rowBase = (long long)blockIdx.x * 128 + wave * 16;

  // ---- per-lane persistent state ----
  int   tsel[8];
  float w3c0[8], w3c1[8], b2v[8];
#pragma unroll
  for (int r = 0; r < 8; ++r) {
    int tv = t[rowBase + r + 8 * lhi];
    tv = tv < 0 ? 0 : tv;                 // clamp_min(0) as in reference
    tsel[r] = tv > 1 ? 1 : tv;
  }
#pragma unroll
  for (int nt = 0; nt < 8; ++nt) {
    int n = nt * 16 + l15;
    w3c0[nt] = W3[n * 2 + 0];
    w3c1[nt] = W3[n * 2 + 1];
    b2v[nt]  = b2[n];
  }

  // ---- precompute c = x @ W1x + b1 (kept in registers, C-layout) ----
  v8f ck[8];
#pragma unroll
  for (int nt = 0; nt < 8; ++nt) {
    float bv = b1[nt * 16 + l15];
#pragma unroll
    for (int j = 0; j < 8; ++j) ck[nt][j] = bv;
  }
  const float* xrow = x + (rowBase + l15) * 128;
#pragma unroll
  for (int kt = 0; kt < 4; ++kt) {
    v16h a = gmem_frag_x(xrow, kt * 32, lhi);
#pragma unroll
    for (int nt = 0; nt < 8; ++nt) {
      v16h b = lds_frag(sW1x + nt * 16 * 128, 128, kt * 32, l15, lhi);
      ck[nt] = WMMA_F16(a, b, ck[nt]);
    }
  }

  float yv[8];
#pragma unroll
  for (int r = 0; r < 8; ++r) yv[r] = 0.f;

  const int nsteps = stepsp[0];
  for (int s = 0; s < nsteps; ++s) {
    // stage y strip (f16, K-padded); same-wave LDS ordering is guaranteed
#pragma unroll
    for (int r = 0; r < 8; ++r)
      ypad[(r + 8 * lhi) * 32 + l15] = (_Float16)yv[r];

    // ---- z1 = y @ W1y + c  (one K=32 WMMA per N-tile, C := ck) ----
    v8f az[8];
    {
      v16h a = lds_frag(ypad, 32, 0, l15, lhi);
#pragma unroll
      for (int nt = 0; nt < 8; ++nt) {
        v16h b = lds_frag(sW1y + nt * 16 * 32, 32, 0, l15, lhi);
        az[nt] = WMMA_F16(a, b, ck[nt]);
      }
    }

    // ---- h1 = relu(z1): pack sign mask, stage f16 strip ----
    unsigned m1lo = 0, m1hi = 0;
#pragma unroll
    for (int nt = 0; nt < 8; ++nt) {
#pragma unroll
      for (int r = 0; r < 8; ++r) {
        float z = az[nt][r];
        unsigned bit = (z > 0.f) ? 1u : 0u;
        unsigned sh  = (unsigned)((nt & 3) * 8 + r);
        if (nt < 4) m1lo |= bit << sh; else m1hi |= bit << sh;
        stg[(r + 8 * lhi) * 128 + nt * 16 + l15] = (_Float16)(z > 0.f ? z : 0.f);
      }
    }

    // ---- z2 = h1 @ W2 + b2 ----
    v8f a2[8];
#pragma unroll
    for (int nt = 0; nt < 8; ++nt) {
#pragma unroll
      for (int j = 0; j < 8; ++j) a2[nt][j] = b2v[nt];
    }
#pragma unroll
    for (int kt = 0; kt < 4; ++kt) {
      v16h a = lds_frag(stg, 128, kt * 32, l15, lhi);
#pragma unroll
      for (int nt = 0; nt < 8; ++nt) {
        v16h b = lds_frag(sW2b + nt * 16 * 128, 128, kt * 32, l15, lhi);
        a2[nt] = WMMA_F16(a, b, a2[nt]);
      }
    }

    // ---- gz2 = (z2>0) * W3[:, t]  (no forward e needed) ----
#pragma unroll
    for (int nt = 0; nt < 8; ++nt) {
#pragma unroll
      for (int r = 0; r < 8; ++r) {
        float g = (a2[nt][r] > 0.f) ? (tsel[r] ? w3c1[nt] : w3c0[nt]) : 0.f;
        stg[(r + 8 * lhi) * 128 + nt * 16 + l15] = (_Float16)g;
      }
    }

    // ---- g1 = gz2 @ W2^T ----
    v8f g1[8];
#pragma unroll
    for (int nt = 0; nt < 8; ++nt) {
      v8f zz = {0.f, 0.f, 0.f, 0.f, 0.f, 0.f, 0.f, 0.f};
      g1[nt] = zz;
    }
#pragma unroll
    for (int kt = 0; kt < 4; ++kt) {
      v16h a = lds_frag(stg, 128, kt * 32, l15, lhi);
#pragma unroll
      for (int nt = 0; nt < 8; ++nt) {
        v16h b = lds_frag(sW2t + nt * 16 * 128, 128, kt * 32, l15, lhi);
        g1[nt] = WMMA_F16(a, b, g1[nt]);
      }
    }

    // ---- gz1 = g1 * (z1>0), stage f16 ----
#pragma unroll
    for (int nt = 0; nt < 8; ++nt) {
#pragma unroll
      for (int r = 0; r < 8; ++r) {
        unsigned sh  = (unsigned)((nt & 3) * 8 + r);
        unsigned bit = ((nt < 4 ? m1lo : m1hi) >> sh) & 1u;
        stg[(r + 8 * lhi) * 128 + nt * 16 + l15] =
            bit ? (_Float16)g1[nt][r] : (_Float16)0.f;
      }
    }

    // ---- g_y = gz1 @ W1y^T  (N = 16 -> single N-tile) ----
    v8f gy = {0.f, 0.f, 0.f, 0.f, 0.f, 0.f, 0.f, 0.f};
#pragma unroll
    for (int kt = 0; kt < 4; ++kt) {
      v16h a = lds_frag(stg, 128, kt * 32, l15, lhi);
      v16h b = lds_frag(sW1yT, 128, kt * 32, l15, lhi);
      gy = WMMA_F16(a, b, gy);
    }

    // ---- y -= LR * g_y  (each (lane,r) uniquely owns y[m][n]) ----
#pragma unroll
    for (int r = 0; r < 8; ++r) yv[r] -= LRATE * gy[r];
  }

  // ---- write y out: row = rowBase + r + 8*lhi, col = l15 ----
#pragma unroll
  for (int r = 0; r < 8; ++r)
    out[(rowBase + r + 8 * lhi) * 16 + l15] = yv[r];
}

extern "C" void kernel_launch(void* const* d_in, const int* in_sizes, int n_in,
                              void* d_out, int out_size, void* d_ws, size_t ws_size,
                              hipStream_t stream) {
  const float* x     = (const float*)d_in[0];
  const int*   t     = (const int*)d_in[1];
  const float* W1    = (const float*)d_in[2];
  const float* b1    = (const float*)d_in[3];
  const float* W2    = (const float*)d_in[4];
  const float* b2    = (const float*)d_in[5];
  const float* W3    = (const float*)d_in[6];
  /* d_in[7] = b3 (unused: gradient wrt y does not see b3) */
  const int*   steps = (const int*)d_in[8];
  float* out = (float*)d_out;

  const int B      = in_sizes[0] / 128;  // rows of x
  const int blocks = B / 128;            // 128 rows per workgroup

  hipFuncSetAttribute((const void*)JointEBM_kernel,
                      hipFuncAttributeMaxDynamicSharedMemorySize, SMEM_BYTES);
  JointEBM_kernel<<<blocks, 256, SMEM_BYTES, stream>>>(x, t, W1, b1, W2, b2, W3,
                                                       steps, out);
}
